// GraphStructureLearner_37297495998603
// MI455X (gfx1250) — matrix operand order repaired and checked by reading, not compiled, not run on hardware
//
#include <hip/hip_runtime.h>
#include <hip/hip_bf16.h>
#include <math.h>

// Problem constants (match reference)
#define NN 10000
#define FF 128
#define DD 64
#define EE 320000

typedef __attribute__((ext_vector_type(16))) _Float16 v16h;
typedef __attribute__((ext_vector_type(8)))  _Float16 v8h;
typedef __attribute__((ext_vector_type(8)))  float    v8f;

// ---------------------------------------------------------------------------
// Workspace layout (bytes, all offsets 32B-aligned given 256B-aligned base):
//   deg/dinv : NN floats            (40,000 B)
//   h        : NN*DD floats         (2,560,000 B)
//   df       : NN*DD floats         (2,560,000 B)
//   de1h     : NN*DD _Float16       (1,280,000 B)
//   de2h     : NN*DD _Float16       (1,280,000 B)
// total ~7.4 MB
// ---------------------------------------------------------------------------

// ---- init: deg = 1.0 (self-loop weight), df = 0 ---------------------------
__global__ void init_kernel(float* deg, float* df) {
    int gid = blockIdx.x * blockDim.x + threadIdx.x;
    if (gid < NN) deg[gid] = 1.0f;
    if (gid < NN * DD) df[gid] = 0.0f;
}

// ---- deg[dst] += ew ---------------------------------------------------------
__global__ void degree_kernel(const int* __restrict__ edges,
                              const float* __restrict__ ew,
                              float* deg) {
    int e = blockIdx.x * blockDim.x + threadIdx.x;
    if (e >= EE) return;
    int dst = edges[EE + e];
    atomicAdd(&deg[dst], ew[e]);
}

// ---- dinv = rsqrt(deg) (deg >= 1 always due to self-loop) ------------------
__global__ void dinv_kernel(float* deg) {
    int i = blockIdx.x * blockDim.x + threadIdx.x;
    if (i >= NN) return;
    float d = deg[i];
    deg[i] = (d > 0.0f) ? rsqrtf(d) : 0.0f;
}

// ---- h = x @ W_g^T   [NN,FF] x [DD,FF] -> [NN,DD] --------------------------
__global__ void lin_g_kernel(const float* __restrict__ x,
                             const float* __restrict__ Wg,
                             float* __restrict__ h) {
    int gid = blockIdx.x * blockDim.x + threadIdx.x;
    if (gid >= NN * DD) return;
    int n = gid / DD, d = gid % DD;
    const float* xr = x + n * FF;
    const float* wr = Wg + d * FF;
    float s = 0.0f;
#pragma unroll 8
    for (int f = 0; f < FF; ++f) s = fmaf(xr[f], wr[f], s);
    h[gid] = s;
}

// ---- scatter: df[dst] += dinv[src]*ew*dinv[dst] * h[src] -------------------
__global__ void scatter_kernel(const int* __restrict__ edges,
                               const float* __restrict__ ew,
                               const float* __restrict__ dinv,
                               const float* __restrict__ h,
                               float* df) {
    long long gid = (long long)blockIdx.x * blockDim.x + threadIdx.x;
    if (gid >= (long long)EE * DD) return;
    int e = (int)(gid >> 6);
    int d = (int)(gid & 63);
    int src = edges[e];
    int dst = edges[EE + e];
    float norm = dinv[src] * ew[e] * dinv[dst];
    atomicAdd(&df[dst * DD + d], norm * h[src * DD + d]);
}

// ---- fixup: add self-loop message + bias -----------------------------------
__global__ void fixup_kernel(const float* __restrict__ dinv,
                             const float* __restrict__ h,
                             const float* __restrict__ bg,
                             float* df) {
    int gid = blockIdx.x * blockDim.x + threadIdx.x;
    if (gid >= NN * DD) return;
    int n = gid / DD, k = gid % DD;
    float di = dinv[n];
    df[gid] += di * di * h[gid] + bg[k];
}

// ---- de1 = tanh(df@W1^T+b1), de2 = tanh(df@W2^T+b2), stored as f16 ---------
__global__ void de_kernel(const float* __restrict__ df,
                          const float* __restrict__ W1, const float* __restrict__ b1,
                          const float* __restrict__ W2, const float* __restrict__ b2,
                          _Float16* __restrict__ de1h, _Float16* __restrict__ de2h) {
    int gid = blockIdx.x * blockDim.x + threadIdx.x;
    if (gid >= NN * DD) return;
    int n = gid / DD, d = gid % DD;
    const float* fr = df + n * DD;
    const float* w1 = W1 + d * DD;
    const float* w2 = W2 + d * DD;
    float s1 = b1[d], s2 = b2[d];
#pragma unroll 8
    for (int k = 0; k < DD; ++k) {
        float f = fr[k];
        s1 = fmaf(f, w1[k], s1);
        s2 = fmaf(f, w2[k], s2);
    }
    de1h[gid] = (_Float16)tanhf(s1);
    de2h[gid] = (_Float16)tanhf(s2);
}

// ---------------------------------------------------------------------------
// Big kernel: Et = relu(tanh(de1@de2^T - de2@de1^T)) via V_WMMA_F32_16X16X32_F16
// One wave per 16-row strip of up to 4 16x16 output tiles; A-fragments reused across 4 consecutive
// J tiles. All tile indices forced scalar via readfirstlane so the tile
// control flow is SALU branches and EXEC stays all-1s around WMMA.
//
// Fragment layouts per CDNA5 ISA 7.12.2 (wave32):
//  A (16x32 f16, M x K):  lane L: M = L%16, half = L/16;
//     v16h elems 0..7  = K = kc + half*8 + 0..7
//     v16h elems 8..15 = K = kc + 16 + half*8 + 0..7
//  B (32x16 f16, K x N):  lane L: N = L%16, half = L/16;
//     v16h elems 0..7  = K = kc + half*8 + 0..7, elems 8..15 = K = kc + 16 + half*8 + 0..7
//     (row loads, because B = (row-major de)^T: column n of B is row n of de)
//  D (16x16 f32): VGPR r -> M = r + 8*(L/16), N = L%16
// ---------------------------------------------------------------------------
#define JTILES 625            /* 10000/16 */
#define JGROUP 4
#define NJG    157            /* ceil(625/4) */
#define NTASK  (625 * NJG)

__device__ inline v16h load_a_frag(const _Float16* __restrict__ base,
                                   int row0, int kc, int lane) {
    int m = lane & 15, hf = lane >> 4;
    const _Float16* p = base + (row0 + m) * DD + kc + hf * 8;
    v8h lo = *(const v8h*)(p);
    v8h hi = *(const v8h*)(p + 16);
    v16h r;
#pragma unroll
    for (int i = 0; i < 8; ++i) { r[i] = lo[i]; r[i + 8] = hi[i]; }
    return r;
}

__device__ inline v16h load_b_frag(const _Float16* __restrict__ base,
                                   int col0, int kc, int lane) {
    int n = lane & 15, hf = lane >> 4;
    const _Float16* p = base + (col0 + n) * DD + kc + hf * 8;
    v8h lo = *(const v8h*)(p);
    v8h hi = *(const v8h*)(p + 16);
    v16h r;
#pragma unroll
    for (int i = 0; i < 8; ++i) { r[i] = lo[i]; r[i + 8] = hi[i]; }
    return r;
}

// relu(tanh(x)) branch-free: tanh(x) = 1 - 2/(e^{2x}+1).
// e->inf => 1 ; e->0 => -1 -> relu 0. Straight-line v_exp/v_rcp, no branches.
__device__ inline float relu_tanh(float x) {
    float e = __expf(2.0f * x);
    float r = 1.0f - 2.0f * __builtin_amdgcn_rcpf(e + 1.0f);
    return r > 0.0f ? r : 0.0f;
}

__global__ __launch_bounds__(256) void
skew_wmma_kernel(const _Float16* __restrict__ de1,
                 const _Float16* __restrict__ de2,
                 float* __restrict__ out) {
    const int lane = threadIdx.x & 31;
    // task is wave-uniform: pin it to an SGPR so guards compile to s_cbranch
    const int task =
        __builtin_amdgcn_readfirstlane(blockIdx.x * 8 + (threadIdx.x >> 5));
    if (task >= NTASK) return;

    const int it = task / NJG;
    const int jg = task % NJG;
    const int i0 = it * 16;

    // A-side fragments for tile-row I (reused over J tiles)
    v16h a1k0 = load_a_frag(de1, i0, 0, lane);
    v16h a1k1 = load_a_frag(de1, i0, 32, lane);
    v16h a2k0 = load_a_frag(de2, i0, 0, lane);
    v16h a2k1 = load_a_frag(de2, i0, 32, lane);

    const int hf = lane >> 4;
    const int colLane = lane & 15;

#pragma unroll
    for (int j = 0; j < JGROUP; ++j) {
        const int jt = __builtin_amdgcn_readfirstlane(jg * JGROUP + j);
        if (jt >= JTILES) break;                     // scalar branch
        const int j0 = jt * 16;

        v16h b2k0 = load_b_frag(de2, j0, 0, lane);
        v16h b2k1 = load_b_frag(de2, j0, 32, lane);
        v16h b1k0 = load_b_frag(de1, j0, 0, lane);
        v16h b1k1 = load_b_frag(de1, j0, 32, lane);

        v8f acc1 = {};
        v8f acc2 = {};
        // acc1 = de1_I @ de2_J^T     (K = 0..63 in two K32 chunks)
        acc1 = __builtin_amdgcn_wmma_f32_16x16x32_f16(false, a1k0, false, b2k0,
                                                      (short)0, acc1, false, false);
        acc1 = __builtin_amdgcn_wmma_f32_16x16x32_f16(false, a1k1, false, b2k1,
                                                      (short)0, acc1, false, false);
        // acc2 = de2_I @ de1_J^T
        acc2 = __builtin_amdgcn_wmma_f32_16x16x32_f16(false, a2k0, false, b1k0,
                                                      (short)0, acc2, false, false);
        acc2 = __builtin_amdgcn_wmma_f32_16x16x32_f16(false, a2k1, false, b1k1,
                                                      (short)0, acc2, false, false);

        // epilogue: relu(tanh(acc1-acc2)), D layout M = r + 8*hf, N = colLane
        const int col = j0 + colLane;
#pragma unroll
        for (int r = 0; r < 8; ++r) {
            out[(long long)(i0 + r + 8 * hf) * NN + col] =
                relu_tanh(acc1[r] - acc2[r]);
        }
    }
}

// ---------------------------------------------------------------------------
extern "C" void kernel_launch(void* const* d_in, const int* in_sizes, int n_in,
                              void* d_out, int out_size, void* d_ws, size_t ws_size,
                              hipStream_t stream) {
    const float* x   = (const float*)d_in[0];      // [NN, FF]
    const int*   edg = (const int*)  d_in[1];      // [2, EE] (row0=src, row1=dst)
    const float* ew  = (const float*)d_in[2];      // [EE]
    const float* Wg  = (const float*)d_in[3];      // [DD, FF]
    const float* bg  = (const float*)d_in[4];      // [DD]
    const float* W1  = (const float*)d_in[5];      // [DD, DD]
    const float* b1  = (const float*)d_in[6];      // [DD]
    const float* W2  = (const float*)d_in[7];      // [DD, DD]
    const float* b2  = (const float*)d_in[8];      // [DD]
    float* out = (float*)d_out;                    // [NN, NN]

    char* ws = (char*)d_ws;
    float*    deg  = (float*)ws;                         ws += (size_t)NN * sizeof(float);
    float*    h    = (float*)ws;                         ws += (size_t)NN * DD * sizeof(float);
    float*    df   = (float*)ws;                         ws += (size_t)NN * DD * sizeof(float);
    _Float16* de1h = (_Float16*)ws;                      ws += (size_t)NN * DD * sizeof(_Float16);
    _Float16* de2h = (_Float16*)ws;

    const int B = 256;

    init_kernel<<<(NN * DD + B - 1) / B, B, 0, stream>>>(deg, df);
    degree_kernel<<<(EE + B - 1) / B, B, 0, stream>>>(edg, ew, deg);
    dinv_kernel<<<(NN + B - 1) / B, B, 0, stream>>>(deg);
    lin_g_kernel<<<(NN * DD + B - 1) / B, B, 0, stream>>>(x, Wg, h);
    {
        long long tot = (long long)EE * DD;
        scatter_kernel<<<(unsigned)((tot + B - 1) / B), B, 0, stream>>>(edg, ew, deg, h, df);
    }
    fixup_kernel<<<(NN * DD + B - 1) / B, B, 0, stream>>>(deg, h, bg, df);
    de_kernel<<<(NN * DD + B - 1) / B, B, 0, stream>>>(df, W1, b1, W2, b2, de1h, de2h);

    // 625x157 wave-tasks, 8 waves (256 threads) per block
    skew_wmma_kernel<<<(NTASK + 7) / 8, B, 0, stream>>>(de1h, de2h, out);
}
